// ScorePredictor_2d_5480378269726
// MI455X (gfx1250) — matrix operand order repaired and checked by reading, not compiled, not run on hardware
//
#include <hip/hip_runtime.h>

// CDNA5 / gfx1250, wave32. Fused edge-score predictor:
//   data = [x[src] | x[dst] | e_f | e_b]  (E x 512 fp32)
//   h    = relu(data @ W1 + b1)           (E x 64)
//   out  = h @ W2 + b2                    (E x 1)
// GEMM1 on v_wmma_f32_16x16x32_bf16 (fp32 accumulate). W1 staged once per
// block into the 320KB WGP LDS as transposed bf16 (padded rows -> conflict-
// free B reads). Each wave owns 2 x 16-edge tiles so every B ds_load feeds
// two WMMAs. fp32->bf16 via v_perm_b32. Seg loop rolled (bounded VGPRs, no
// spills); segment pointers typed address_space(1) so the selected pointer
// still lowers to global_load_b128 (LOADcnt only), not flat_load (LOAD+DS).

typedef __attribute__((ext_vector_type(16))) __bf16          v16bf;
typedef __attribute__((ext_vector_type(8)))  float           v8f;
typedef __attribute__((ext_vector_type(8)))  unsigned int    v8u;
typedef __attribute__((ext_vector_type(4)))  float           f4;

// Explicit global-address-space pointer types (keep loads on the GLOBAL path).
typedef const __attribute__((address_space(1))) float  GF;
typedef const __attribute__((address_space(1))) f4     GF4;

#define D_IN    128
#define H_HID   64
#define KTOT    512
#define KPAD    520      // 512 + 8 ushort pad -> conflict-free B fragment reads
#define TILE_M  16
#define MT      2        // edge tiles per wave iteration

__device__ __forceinline__ unsigned short f2bf(float f) {
  unsigned int u = __builtin_bit_cast(unsigned int, f);
  unsigned int r = u + 0x7FFFu + ((u >> 16) & 1u);
  return (unsigned short)(r >> 16);
}

// Pack two fp32 into two truncated bf16 with one v_perm_b32.
__device__ __forceinline__ unsigned int pk2(float lo, float hi) {
  return __builtin_amdgcn_perm(__builtin_bit_cast(unsigned int, hi),
                               __builtin_bit_cast(unsigned int, lo),
                               0x07060302u);
}

struct Q2 { uint4 a, b; };   // 32-byte carrier for a B fragment (16 bf16)

__global__ __launch_bounds__(128, 4)
void edge_score_wmma_kernel(const float* __restrict__ x,
                            const float* __restrict__ ef,
                            const float* __restrict__ eb,
                            const int*   __restrict__ src,
                            const int*   __restrict__ dst,
                            const float* __restrict__ W1,
                            const float* __restrict__ b1,
                            const float* __restrict__ W2,
                            const float* __restrict__ b2,
                            float* __restrict__ out,
                            int E) {
  __shared__ unsigned short w1t[H_HID * KPAD];  // bf16 W1 transposed [n][k]
  __shared__ float b1s[H_HID];
  __shared__ float w2s[H_HID];

  for (int i = threadIdx.x; i < KTOT * H_HID; i += blockDim.x) {
    int k = i >> 6;
    int n = i & 63;
    w1t[n * KPAD + k] = f2bf(W1[i]);
  }
  if (threadIdx.x < H_HID) {
    b1s[threadIdx.x] = b1[threadIdx.x];
    w2s[threadIdx.x] = W2[threadIdx.x];
  }
  __syncthreads();

  const float b2v = b2[0];

  // Global-AS views of the feature arrays.
  GF* xg  = (GF*)x;
  GF* efg = (GF*)ef;
  GF* ebg = (GF*)eb;

  const int lane  = threadIdx.x & 31;
  const int n0    = lane & 15;    // A: row-in-tile / B,C: column-in-tile
  const int half  = lane >> 4;    // K sub-range (A/B), M half (C)
  const int wavesPerBlock = blockDim.x >> 5;
  const int gwave  = blockIdx.x * wavesPerBlock + (threadIdx.x >> 5);
  const int nwaves = gridDim.x * wavesPerBlock;
  const int numTiles = (E + TILE_M - 1) / TILE_M;
  const int numPairs = (numTiles + MT - 1) / MT;

  #pragma unroll 1
  for (int pair = gwave; pair < numPairs; pair += nwaves) {
    const int tile0 = pair * MT;

    // Per-tile segment base pointers (4 segments of the 512-wide row).
    GF* sp[MT][4];
    #pragma unroll
    for (int mt = 0; mt < MT; ++mt) {
      const int e  = (tile0 + mt) * TILE_M + n0;
      const int ec = (e < E) ? e : (E - 1);
      const int se = src[ec];
      const int de = dst[ec];
      sp[mt][0] = xg  + (size_t)se * D_IN;
      sp[mt][1] = xg  + (size_t)de * D_IN;
      sp[mt][2] = efg + (size_t)ec * D_IN;
      sp[mt][3] = ebg + (size_t)ec * D_IN;
    }

    // Prefetch next iteration's streaming rows.
    const int npair = pair + nwaves;
    if (npair < numPairs) {
      const size_t ne0 = (size_t)(npair * MT * TILE_M + n0) * D_IN;
      __builtin_prefetch(ef + ne0, 0, 3);
      __builtin_prefetch(eb + ne0, 0, 3);
      __builtin_prefetch(ef + ne0 + (size_t)TILE_M * D_IN, 0, 3);
      __builtin_prefetch(eb + ne0 + (size_t)TILE_M * D_IN, 0, 3);
    }

    v8f cacc[MT][4];
    #pragma unroll
    for (int mt = 0; mt < MT; ++mt)
      #pragma unroll
      for (int t = 0; t < 4; ++t) cacc[mt][t] = (v8f){0,0,0,0,0,0,0,0};

    // K loop: outer over 4 segments (rolled -> bounded VGPR live set; the
    // cndmask-selected pointer stays address_space(1) -> global_load path),
    // inner over 4 k-steps of 32 (unrolled; constant offsets).
    #pragma unroll 1
    for (int seg = 0; seg < 4; ++seg) {
      GF* pa = sp[0][0];
      if (seg == 1) pa = sp[0][1];
      if (seg == 2) pa = sp[0][2];
      if (seg == 3) pa = sp[0][3];
      GF* pb = sp[1][0];
      if (seg == 1) pb = sp[1][1];
      if (seg == 2) pb = sp[1][2];
      if (seg == 3) pb = sp[1][3];
      const unsigned short* bseg = w1t + seg * 4 * 32 + half * 8;

      #pragma unroll
      for (int kq = 0; kq < 4; ++kq) {
        // ---- A fragments (16x32 bf16, ISA layout):
        //   lane(0-15): M=lane, K = 0..7 (V0-3), 16..23 (V4-7); lanes 16-31: +8
        v16bf afr[MT];
        #pragma unroll
        for (int mt = 0; mt < MT; ++mt) {
          GF* p = (mt == 0 ? pa : pb) + kq * 32 + half * 8;
          const f4 f0 = *(GF4*)(p);
          const f4 f1 = *(GF4*)(p + 4);
          const f4 f2 = *(GF4*)(p + 16);
          const f4 f3 = *(GF4*)(p + 20);
          v8u au;
          au[0] = pk2(f0.x, f0.y); au[1] = pk2(f0.z, f0.w);
          au[2] = pk2(f1.x, f1.y); au[3] = pk2(f1.z, f1.w);
          au[4] = pk2(f2.x, f2.y); au[5] = pk2(f2.z, f2.w);
          au[6] = pk2(f3.x, f3.y); au[7] = pk2(f3.z, f3.w);
          afr[mt] = __builtin_bit_cast(v16bf, au);
        }

        // ---- B fragments (32x16 bf16): lane(0-15): N=lane, K=0..7,16..23;
        //      lanes 16-31: K=8..15,24..31. Each feeds both M tiles.
        const unsigned short* brow = bseg + kq * 32;
        #pragma unroll
        for (int t = 0; t < 4; ++t) {
          const unsigned short* bp = brow + (size_t)(n0 + 16 * t) * KPAD;
          Q2 q;
          q.a = *(const uint4*)(bp);
          q.b = *(const uint4*)(bp + 16);
          const v16bf bf = __builtin_bit_cast(v16bf, q);
          cacc[0][t] = __builtin_amdgcn_wmma_f32_16x16x32_bf16(
              false, afr[0], false, bf, (short)0, cacc[0][t], false, false);
          cacc[1][t] = __builtin_amdgcn_wmma_f32_16x16x32_bf16(
              false, afr[1], false, bf, (short)0, cacc[1][t], false, false);
        }
      }
    }

    // ---- Epilogue: relu(h + b1) . W2 reduced over hidden dim.
    // C layout: vgpr r, lane l -> M = r + 8*(l>>4), N = (l&15) + 16*t.
    float bb[4], ww[4];
    #pragma unroll
    for (int t = 0; t < 4; ++t) {
      bb[t] = b1s[n0 + 16 * t];
      ww[t] = w2s[n0 + 16 * t];
    }
    #pragma unroll
    for (int mt = 0; mt < MT; ++mt) {
      float acc[8];
      #pragma unroll
      for (int r = 0; r < 8; ++r) {
        float s = 0.f;
        #pragma unroll
        for (int t = 0; t < 4; ++t)
          s += fmaxf(cacc[mt][t][r] + bb[t], 0.f) * ww[t];
        s += __shfl_xor(s, 1, 32);
        s += __shfl_xor(s, 2, 32);
        s += __shfl_xor(s, 4, 32);
        s += __shfl_xor(s, 8, 32);
        acc[r] = s;
      }
      if (n0 == 0) {
        #pragma unroll
        for (int r = 0; r < 8; ++r) {
          const int oe = (tile0 + mt) * TILE_M + half * 8 + r;
          if (oe < E) out[oe] = acc[r] + b2v;
        }
      }
    }
  }
}

extern "C" void kernel_launch(void* const* d_in, const int* in_sizes, int n_in,
                              void* d_out, int out_size, void* d_ws, size_t ws_size,
                              hipStream_t stream) {
  const float* x  = (const float*)d_in[0];
  const float* ef = (const float*)d_in[1];
  const float* eb = (const float*)d_in[2];
  const int*   src = (const int*)d_in[3];
  const int*   dst = (const int*)d_in[4];
  const float* W1 = (const float*)d_in[5];
  const float* b1 = (const float*)d_in[6];
  const float* W2 = (const float*)d_in[7];
  const float* b2 = (const float*)d_in[8];
  float* out = (float*)d_out;

  const int E = in_sizes[3];

  dim3 grid(1024), block(128);
  hipLaunchKernelGGL(edge_score_wmma_kernel, grid, block, 0, stream,
                     x, ef, eb, src, dst, W1, b1, W2, b2, out, E);
}
